// LSTM_Gumbel_Encoder_Decoder_16243566314016
// MI455X (gfx1250) — compile-verified, matrix operand order checked by reading, not run on hardware
//
#include <hip/hip_runtime.h>
#include <hip/hip_bf16.h>
#include <cstdint>
#include <cstddef>

// ---------------------------------------------------------------------------
// Problem constants (from reference setup_inputs)
// ---------------------------------------------------------------------------
#define B_    32
#define L_    512
#define V_    32000
#define E_    256
#define H_    256      // encoder hidden per direction
#define DH_   512      // decoder hidden (2H)
#define G_    1024     // 4H encoder gate width
#define GD_   2048     // 4Dh decoder gate width
#define TDEC  63       // max_len-1 decode steps
#define HEADP 112      // head rows 103 padded to 16

typedef __attribute__((ext_vector_type(16))) _Float16 v16h;
typedef __attribute__((ext_vector_type(8)))  float    v8f;

#define WMMA(a, b, c) \
  __builtin_amdgcn_wmma_f32_16x16x32_f16(false, (a), false, (b), (short)0, (c), false, false)

static __device__ __forceinline__ float sig_(float x) { return 1.0f / (1.0f + expf(-x)); }

// ---------------------------------------------------------------------------
// Utility kernels
// ---------------------------------------------------------------------------
__global__ void k_zero_f16(_Float16* p, int n) {
  int i = blockIdx.x * 256 + threadIdx.x;
  if (i < n) p[i] = (_Float16)0.0f;
}

__global__ void k_cast(_Float16* d, const float* s, int n) {
  int i = blockIdx.x * 256 + threadIdx.x;
  if (i < n) d[i] = (_Float16)s[i];
}

// d[r*lddst + c] = s[r*ldsrc + c]
__global__ void k_copy2d_cast(_Float16* d, const float* s, int rows, int cols,
                              int lddst, int ldsrc) {
  int i = blockIdx.x * 256 + threadIdx.x;
  if (i >= rows * cols) return;
  int r = i / cols, c = i % cols;
  d[(size_t)r * lddst + c] = (_Float16)s[(size_t)r * ldsrc + c];
}

__global__ void k_bias_sum(float* o, const float* a, const float* b, int n) {
  int i = blockIdx.x * 256 + threadIdx.x;
  if (i < n) o[i] = a[i] + b[i];
}

// xs0[(t*B+b)*E + e] = emb16[x[b*L+t]*E + e]
__global__ void k_embed_seq(const int* x, const _Float16* emb16, _Float16* xs0) {
  long i = (long)blockIdx.x * 256 + threadIdx.x;
  if (i >= (long)L_ * B_ * E_) return;
  int e = (int)(i % E_);
  long tb = i / E_;
  int b = (int)(tb % B_);
  int t = (int)(tb / B_);
  int tok = x[b * L_ + t];
  xs0[i] = emb16[(size_t)tok * E_ + e];
}

__global__ void k_embed_tok(const int* tok, const _Float16* emb16, _Float16* e16) {
  int i = blockIdx.x * 256 + threadIdx.x;
  if (i >= B_ * E_) return;
  int b = i / E_, j = i % E_;
  e16[i] = emb16[(size_t)tok[b] * E_ + j];
}

// ---------------------------------------------------------------------------
// Generic WMMA GEMM: C[M,N] = A[M,K] @ W[N,K]^T (+bias[N]).  f16 in, f32 out.
// grid = (N/16, M/16), block = 32 (one wave per 16x16 C tile).
// Per documented gfx1250 C-layout: lane l -> N = tile*16 + (l&15),
// VGPR r -> M = tile*16 + (l>>4)*8 + r.
// ---------------------------------------------------------------------------
__global__ void k_wmma_gemm_nt(const _Float16* __restrict__ A, int lda,
                               const _Float16* __restrict__ W,
                               const float* __restrict__ bias,
                               float* __restrict__ C, int ldc, int K) {
  int lane = threadIdx.x;
  int tn = blockIdx.x, tm = blockIdx.y;
  int row = lane & 15, half = lane >> 4;
  const _Float16* Ar = A + (size_t)(tm * 16 + row) * lda + half * 16;
  const _Float16* Wr = W + (size_t)(tn * 16 + row) * K + half * 16;
  v8f acc = {};
  for (int k = 0; k < K; k += 32) {
    __builtin_prefetch(Wr + k + 128, 0, 1);
    v16h a = *(const v16h*)(Ar + k);
    v16h b = *(const v16h*)(Wr + k);
    acc = WMMA(a, b, acc);
  }
  int n = tn * 16 + (lane & 15);
  int mb = tm * 16 + half * 8;
  float bv = bias ? bias[n] : 0.0f;
#pragma unroll
  for (int r = 0; r < 8; ++r) C[(size_t)(mb + r) * ldc + n] = acc[r] + bv;
}

// ---------------------------------------------------------------------------
// Persistent encoder recurrence (one direction of one layer).
// 512 threads = 16 waves; wave w owns hidden cols [16w,16w+16) and all 4 gates
// for those cols, so activations are wave-local.  c stays in registers for the
// whole 512-step scan; h ping-pongs in LDS (f16).
// Gbuf = x@W_ih^T + b_ih + b_hh, precomputed by the big WMMA GEMM.
// ---------------------------------------------------------------------------
__global__ __launch_bounds__(512) void k_enc_recur(
    const float* __restrict__ Gbuf,       // [L][B][1024]
    const _Float16* __restrict__ Whh,     // [1024][256]
    _Float16* __restrict__ xs_out,        // [L][B][512], write cols dirOff..+255
    float* __restrict__ hfin,             // [32][512] decoder h init
    float* __restrict__ cfin,             // [32][512] decoder c init
    int reverse, int dirOff) {
  __shared__ _Float16 hsh[2][B_][H_];
  int tid = threadIdx.x;
  int wave = tid >> 5, lane = tid & 31;
  int col = lane & 15, half = lane >> 4;
  int hcol = wave * 16 + col;

  for (int i = tid; i < B_ * H_; i += 512) ((_Float16*)hsh[0])[i] = (_Float16)0.0f;
  __syncthreads();

  float creg[16];
#pragma unroll
  for (int i = 0; i < 16; ++i) creg[i] = 0.0f;

  for (int s = 0; s < L_; ++s) {
    int t = reverse ? (L_ - 1 - s) : s;
    int cur = s & 1, nxt = cur ^ 1;
    v8f acc[8] = {};  // [gate*2 + mtile]
#pragma unroll
    for (int kk = 0; kk < H_; kk += 32) {
      v16h a0 = *(const v16h*)&hsh[cur][col][kk + half * 16];
      v16h a1 = *(const v16h*)&hsh[cur][16 + col][kk + half * 16];
#pragma unroll
      for (int g = 0; g < 4; ++g) {
        const _Float16* wr = Whh + (size_t)(g * H_ + wave * 16 + col) * H_ + kk + half * 16;
        v16h b = *(const v16h*)wr;
        acc[g * 2 + 0] = WMMA(a0, b, acc[g * 2 + 0]);
        acc[g * 2 + 1] = WMMA(a1, b, acc[g * 2 + 1]);
      }
    }
#pragma unroll
    for (int mt = 0; mt < 2; ++mt) {
#pragma unroll
      for (int r = 0; r < 8; ++r) {
        int m = mt * 16 + half * 8 + r;
        const float* gp = Gbuf + ((size_t)t * B_ + m) * G_ + hcol;
        float gi = acc[0 + mt][r] + gp[0];
        float gf = acc[2 + mt][r] + gp[H_];
        float gg = acc[4 + mt][r] + gp[2 * H_];
        float go = acc[6 + mt][r] + gp[3 * H_];
        float cc = sig_(gf) * creg[mt * 8 + r] + sig_(gi) * tanhf(gg);
        float hh = sig_(go) * tanhf(cc);
        creg[mt * 8 + r] = cc;
        hsh[nxt][m][hcol] = (_Float16)hh;
        xs_out[((size_t)t * B_ + m) * (2 * H_) + dirOff + hcol] = (_Float16)hh;
        if (s == L_ - 1) {
          hfin[(size_t)m * DH_ + dirOff + hcol] = hh;
          cfin[(size_t)m * DH_ + dirOff + hcol] = cc;
        }
      }
    }
    __syncthreads();
  }
}

// ---------------------------------------------------------------------------
// Decoder LSTM cell: gates = Xin@Wih^T + Hprev@Whh^T + bsum, then activations.
// 1024 threads = 32 waves; wave w owns hidden cols [16w,16w+16) and its 4 gates.
// ---------------------------------------------------------------------------
__global__ __launch_bounds__(1024) void k_dec_cell(
    const _Float16* __restrict__ Xin, int Kin,    // [32][Kin]
    const _Float16* __restrict__ Hprev,           // [32][512]
    const _Float16* __restrict__ Wih,             // [2048][Kin]
    const _Float16* __restrict__ Whh,             // [2048][512]
    const float* __restrict__ bsum,               // [2048]
    float* __restrict__ Cst,                      // [32][512] in/out
    float* __restrict__ Hout32,                   // [32][512]
    _Float16* __restrict__ Hout16) {              // [32][512]
  int tid = threadIdx.x;
  int wave = tid >> 5, lane = tid & 31;
  int col = lane & 15, half = lane >> 4;
  int hcol = wave * 16 + col;
  v8f acc[8] = {};
  for (int k = 0; k < Kin; k += 32) {
    v16h a0 = *(const v16h*)(Xin + (size_t)col * Kin + k + half * 16);
    v16h a1 = *(const v16h*)(Xin + (size_t)(16 + col) * Kin + k + half * 16);
#pragma unroll
    for (int g = 0; g < 4; ++g) {
      const _Float16* wr = Wih + (size_t)(g * DH_ + wave * 16 + col) * Kin + k + half * 16;
      __builtin_prefetch(wr + 64, 0, 1);
      v16h b = *(const v16h*)wr;
      acc[g * 2 + 0] = WMMA(a0, b, acc[g * 2 + 0]);
      acc[g * 2 + 1] = WMMA(a1, b, acc[g * 2 + 1]);
    }
  }
  for (int k = 0; k < DH_; k += 32) {
    v16h a0 = *(const v16h*)(Hprev + (size_t)col * DH_ + k + half * 16);
    v16h a1 = *(const v16h*)(Hprev + (size_t)(16 + col) * DH_ + k + half * 16);
#pragma unroll
    for (int g = 0; g < 4; ++g) {
      const _Float16* wr = Whh + (size_t)(g * DH_ + wave * 16 + col) * DH_ + k + half * 16;
      v16h b = *(const v16h*)wr;
      acc[g * 2 + 0] = WMMA(a0, b, acc[g * 2 + 0]);
      acc[g * 2 + 1] = WMMA(a1, b, acc[g * 2 + 1]);
    }
  }
#pragma unroll
  for (int mt = 0; mt < 2; ++mt) {
#pragma unroll
    for (int r = 0; r < 8; ++r) {
      int m = mt * 16 + half * 8 + r;
      float gi = acc[0 + mt][r] + bsum[0 * DH_ + hcol];
      float gf = acc[2 + mt][r] + bsum[1 * DH_ + hcol];
      float gg = acc[4 + mt][r] + bsum[2 * DH_ + hcol];
      float go = acc[6 + mt][r] + bsum[3 * DH_ + hcol];
      float cold = Cst[(size_t)m * DH_ + hcol];
      float cc = sig_(gf) * cold + sig_(gi) * tanhf(gg);
      float hh = sig_(go) * tanhf(cc);
      Cst[(size_t)m * DH_ + hcol] = cc;
      Hout32[(size_t)m * DH_ + hcol] = hh;
      Hout16[(size_t)m * DH_ + hcol] = (_Float16)hh;
    }
  }
}

// ---------------------------------------------------------------------------
// Attention: scores+softmax (block per batch), then context+lg assembly.
// memory is the f16 encoder output xs2 in [L][B][2H] layout.
// ---------------------------------------------------------------------------
__global__ void k_attn(const _Float16* __restrict__ mem, const float* __restrict__ h1,
                       float* __restrict__ attn) {
  int b = blockIdx.x, tid = threadIdx.x;  // 256 threads
  __shared__ float sc[L_];
  __shared__ float red[256];
  const float* hp = h1 + (size_t)b * DH_;
  for (int l = tid; l < L_; l += 256) {
    const _Float16* mp = mem + ((size_t)l * B_ + b) * DH_;
    float s = 0.0f;
    for (int e = 0; e < DH_; ++e) s += (float)mp[e] * hp[e];
    sc[l] = s;
  }
  __syncthreads();
  red[tid] = fmaxf(sc[tid], sc[tid + 256]);
  __syncthreads();
  for (int o = 128; o > 0; o >>= 1) {
    if (tid < o) red[tid] = fmaxf(red[tid], red[tid + o]);
    __syncthreads();
  }
  float mx = red[0];
  __syncthreads();
  red[tid] = expf(sc[tid] - mx) + expf(sc[tid + 256] - mx);
  __syncthreads();
  for (int o = 128; o > 0; o >>= 1) {
    if (tid < o) red[tid] += red[tid + o];
    __syncthreads();
  }
  float inv = 1.0f / red[0];
  for (int l = tid; l < L_; l += 256) attn[(size_t)b * L_ + l] = expf(sc[l] - mx) * inv;
}

__global__ void k_ctx(const _Float16* __restrict__ mem, const float* __restrict__ attn,
                      const float* __restrict__ h1, _Float16* __restrict__ lg) {
  int b = blockIdx.x, tid = threadIdx.x;  // 256 threads
  for (int e = tid; e < DH_; e += 256) {
    float s = 0.0f;
    for (int l = 0; l < L_; ++l)
      s += attn[(size_t)b * L_ + l] * (float)mem[((size_t)l * B_ + b) * DH_ + e];
    lg[(size_t)b * 1024 + DH_ + e] = (_Float16)s;
    lg[(size_t)b * 1024 + e] = (_Float16)h1[(size_t)b * DH_ + e];
  }
}

__global__ void k_critic(const float* __restrict__ h1, const float* __restrict__ cw,
                         const float* __restrict__ cb, float* __restrict__ outp, int step) {
  int b = blockIdx.x, tid = threadIdx.x;  // 128 threads
  __shared__ float red[128];
  float s = 0.0f;
  for (int e = tid; e < DH_; e += 128) s += h1[(size_t)b * DH_ + e] * cw[e];
  red[tid] = s;
  __syncthreads();
  for (int o = 64; o > 0; o >>= 1) {
    if (tid < o) red[tid] += red[tid + o];
    __syncthreads();
  }
  if (tid == 0) outp[b * TDEC + step] = red[0] + cb[0];
}

// ---------------------------------------------------------------------------
// Adaptive log-softmax pieces
// ---------------------------------------------------------------------------
__global__ void k_head_lse(const float* __restrict__ hl, float* __restrict__ hls) {
  int b = blockIdx.x, tid = threadIdx.x;  // 128 threads, 103 valid cols
  __shared__ float red[128];
  float m = -1e30f;
  for (int v = tid; v < 103; v += 128) m = fmaxf(m, hl[b * HEADP + v]);
  red[tid] = m;
  __syncthreads();
  for (int o = 64; o > 0; o >>= 1) {
    if (tid < o) red[tid] = fmaxf(red[tid], red[tid + o]);
    __syncthreads();
  }
  float mx = red[0];
  __syncthreads();
  float s = 0.0f;
  for (int v = tid; v < 103; v += 128) s += expf(hl[b * HEADP + v] - mx);
  red[tid] = s;
  __syncthreads();
  for (int o = 64; o > 0; o >>= 1) {
    if (tid < o) red[tid] += red[tid + o];
    __syncthreads();
  }
  float lse = mx + logf(red[0]);
  for (int v = tid; v < 103; v += 128) hls[b * HEADP + v] = hl[b * HEADP + v] - lse;
}

__global__ void k_head_probs(const float* __restrict__ hls, float* __restrict__ probs,
                             float* __restrict__ cand_v, int* __restrict__ cand_i, int step) {
  int b = blockIdx.x, tid = threadIdx.x;  // 128 threads
  __shared__ float rv[128];
  __shared__ int ri[128];
  float bv = -1.0f;
  int bi = 0;
  float* dst = probs + ((size_t)b * TDEC + step) * V_;
  for (int v = tid; v < 100; v += 128) {
    float p = expf(hls[b * HEADP + v]);
    dst[v] = p;
    if (p > bv) { bv = p; bi = v; }
  }
  rv[tid] = bv; ri[tid] = bi;
  __syncthreads();
  for (int o = 64; o > 0; o >>= 1) {
    if (tid < o && rv[tid + o] > rv[tid]) { rv[tid] = rv[tid + o]; ri[tid] = ri[tid + o]; }
    __syncthreads();
  }
  if (tid == 0) { cand_v[b * 4 + 0] = rv[0]; cand_i[b * 4 + 0] = ri[0]; }
}

__global__ void k_tail_probs(const float* __restrict__ tl, int ldt, int osz,
                             const float* __restrict__ hls, int hcol,
                             float* __restrict__ probs, int vbase,
                             float* __restrict__ cand_v, int* __restrict__ cand_i,
                             int slot, int step) {
  int b = blockIdx.x, tid = threadIdx.x;  // 256 threads
  __shared__ float red[256];
  __shared__ int ri[256];
  const float* row = tl + (size_t)b * ldt;
  float m = -1e30f;
  int mi = 0;
  for (int v = tid; v < osz; v += 256) {
    float x = row[v];
    if (x > m) { m = x; mi = v; }
  }
  red[tid] = m; ri[tid] = mi;
  __syncthreads();
  for (int o = 128; o > 0; o >>= 1) {
    if (tid < o && red[tid + o] > red[tid]) { red[tid] = red[tid + o]; ri[tid] = ri[tid + o]; }
    __syncthreads();
  }
  float mx = red[0];
  int ami = ri[0];
  __syncthreads();
  float s = 0.0f;
  for (int v = tid; v < osz; v += 256) s += expf(row[v] - mx);
  red[tid] = s;
  __syncthreads();
  for (int o = 128; o > 0; o >>= 1) {
    if (tid < o) red[tid] += red[tid + o];
    __syncthreads();
  }
  float lse = mx + logf(red[0]);
  float base = hls[b * HEADP + hcol];
  float* dst = probs + ((size_t)b * TDEC + step) * V_ + vbase;
  for (int v = tid; v < osz; v += 256) dst[v] = expf(row[v] - lse + base);
  if (tid == 0) {
    cand_v[b * 4 + slot] = expf(mx - lse + base);
    cand_i[b * 4 + slot] = vbase + ami;
  }
}

__global__ void k_finalize(const float* __restrict__ cand_v, const int* __restrict__ cand_i,
                           float* __restrict__ ys, float* __restrict__ vals,
                           int* __restrict__ tok, int step) {
  int b = threadIdx.x;
  if (b >= B_) return;
  float bv = -1.0f;
  int bi = 0;
  for (int s = 0; s < 4; ++s)
    if (cand_v[b * 4 + s] > bv) { bv = cand_v[b * 4 + s]; bi = cand_i[b * 4 + s]; }
  vals[b * TDEC + step] = bv;
  ys[b * 64 + step + 1] = (float)bi;
  tok[b] = bi;
}

__global__ void k_onehot(const int* __restrict__ tok, float* __restrict__ oh, int step) {
  long i = (long)blockIdx.x * 256 + threadIdx.x;
  if (i >= (long)B_ * V_) return;
  int b = (int)(i / V_);
  int v = (int)(i % V_);
  oh[((size_t)b * TDEC + step) * V_ + v] = (v == tok[b]) ? 1.0f : 0.0f;
}

__global__ void k_dec_init(const int* __restrict__ start, float* __restrict__ ys,
                           int* __restrict__ tok) {
  int b = threadIdx.x;
  if (b >= B_) return;
  ys[b * 64] = (float)start[0];
  tok[b] = start[0];
}

// ---------------------------------------------------------------------------
// Host orchestration
// ---------------------------------------------------------------------------
static inline int cdiv(long a, long b) { return (int)((a + b - 1) / b); }

extern "C" void kernel_launch(void* const* d_in, const int* in_sizes, int n_in,
                              void* d_out, int out_size, void* d_ws, size_t ws_size,
                              hipStream_t stream) {
  (void)in_sizes; (void)n_in; (void)out_size; (void)ws_size;

  // ---- inputs (flattened dict order) ----
  int p = 0;
  const int* x = (const int*)d_in[p++];        // [B,L]
  p++;                                         // src_mask (unused)
  p++;                                         // max_len (known: 64)
  const int* start_sym = (const int*)d_in[p++];
  const float* emb = (const float*)d_in[p++];  // [V,E]
  const float *enc_wih[2][2], *enc_whh[2][2], *enc_bih[2][2], *enc_bhh[2][2];
  for (int l = 0; l < 2; ++l)
    for (int d = 0; d < 2; ++d) {
      enc_wih[l][d] = (const float*)d_in[p++];
      enc_whh[l][d] = (const float*)d_in[p++];
      enc_bih[l][d] = (const float*)d_in[p++];
      enc_bhh[l][d] = (const float*)d_in[p++];
    }
  const float *dec_wih[2], *dec_whh[2], *dec_bih[2], *dec_bhh[2];
  for (int l = 0; l < 2; ++l) {
    dec_wih[l] = (const float*)d_in[p++];
    dec_whh[l] = (const float*)d_in[p++];
    dec_bih[l] = (const float*)d_in[p++];
    dec_bhh[l] = (const float*)d_in[p++];
  }
  const float* crit_w = (const float*)d_in[p++];
  const float* crit_b = (const float*)d_in[p++];
  const float* head_w = (const float*)d_in[p++];  // [103,1024]
  const float *proj_w[3], *out_w[3];
  for (int i = 0; i < 3; ++i) {
    proj_w[i] = (const float*)d_in[p++];
    out_w[i]  = (const float*)d_in[p++];
  }

  // ---- workspace carve-up ----
  char* ws = (char*)d_ws;
  size_t off = 0;
  auto alloc = [&](size_t bytes) -> char* {
    char* r = ws + off;
    off = (off + bytes + 255) & ~(size_t)255;
    return r;
  };
  _Float16* emb16 = (_Float16*)alloc((size_t)V_ * E_ * 2);
  _Float16* xs0   = (_Float16*)alloc((size_t)L_ * B_ * E_ * 2);
  _Float16* xs1   = (_Float16*)alloc((size_t)L_ * B_ * DH_ * 2);
  _Float16* xs2   = (_Float16*)alloc((size_t)L_ * B_ * DH_ * 2);   // memory
  float*    Gbuf  = (float*)alloc((size_t)L_ * B_ * G_ * 4);
  _Float16 *ewih16[2][2], *ewhh16[2][2];
  float*    ebsum[2][2];
  for (int l = 0; l < 2; ++l)
    for (int d = 0; d < 2; ++d) {
      int din = (l == 0) ? E_ : DH_;
      ewih16[l][d] = (_Float16*)alloc((size_t)G_ * din * 2);
      ewhh16[l][d] = (_Float16*)alloc((size_t)G_ * H_ * 2);
      ebsum[l][d]  = (float*)alloc((size_t)G_ * 4);
    }
  _Float16 *dwih16[2], *dwhh16[2];
  float* dbsum[2];
  for (int l = 0; l < 2; ++l) {
    int din = (l == 0) ? E_ : DH_;
    dwih16[l] = (_Float16*)alloc((size_t)GD_ * din * 2);
    dwhh16[l] = (_Float16*)alloc((size_t)GD_ * DH_ * 2);
    dbsum[l]  = (float*)alloc((size_t)GD_ * 4);
  }
  _Float16* headw16 = (_Float16*)alloc((size_t)HEADP * 1024 * 2);
  const int hsz[3]  = {64, 16, 4};
  const int hp[3]   = {64, 16, 16};    // proj rows padded
  const int Kp[3]   = {64, 32, 32};    // proj cols / out_w K padded
  const int osz[3]  = {900, 9000, 22000};
  const int oszp[3] = {912, 9008, 22000};
  const int vbase[3] = {100, 1000, 10000};
  _Float16 *projw16[3], *outw16[3], *projf16[3];
  for (int i = 0; i < 3; ++i) {
    projw16[i] = (_Float16*)alloc((size_t)hp[i] * 1024 * 2);
    outw16[i]  = (_Float16*)alloc((size_t)oszp[i] * Kp[i] * 2);
    projf16[i] = (_Float16*)alloc((size_t)B_ * Kp[i] * 2);
  }
  float* h0init = (float*)alloc((size_t)B_ * DH_ * 4);
  float* h1init = (float*)alloc((size_t)B_ * DH_ * 4);
  float* c0st   = (float*)alloc((size_t)B_ * DH_ * 4);
  float* c1st   = (float*)alloc((size_t)B_ * DH_ * 4);
  _Float16* h0f16[2] = {(_Float16*)alloc((size_t)B_ * DH_ * 2),
                        (_Float16*)alloc((size_t)B_ * DH_ * 2)};
  _Float16* h1f16[2] = {(_Float16*)alloc((size_t)B_ * DH_ * 2),
                        (_Float16*)alloc((size_t)B_ * DH_ * 2)};
  float* hstep0 = (float*)alloc((size_t)B_ * DH_ * 4);
  float* hstep1 = (float*)alloc((size_t)B_ * DH_ * 4);
  _Float16* e16 = (_Float16*)alloc((size_t)B_ * E_ * 2);
  float* attn_b = (float*)alloc((size_t)B_ * L_ * 4);
  _Float16* lg16 = (_Float16*)alloc((size_t)B_ * 1024 * 2);
  float* headlog = (float*)alloc((size_t)B_ * HEADP * 4);
  float* headls  = (float*)alloc((size_t)B_ * HEADP * 4);
  float* projf32 = (float*)alloc((size_t)B_ * 64 * 4);
  float* taillog = (float*)alloc((size_t)B_ * 22016 * 4);
  float* cand_v  = (float*)alloc((size_t)B_ * 4 * 4);
  int*   cand_i  = (int*)alloc((size_t)B_ * 4 * 4);
  int*   tok     = (int*)alloc((size_t)B_ * 4);

  // ---- output carve-up (return order: ys, values, critics, probs, one_hots)
  float* out = (float*)d_out;
  float* ys_o   = out;
  float* val_o  = out + (size_t)B_ * 64;
  float* crit_o = val_o + (size_t)B_ * TDEC;
  float* prob_o = crit_o + (size_t)B_ * TDEC;
  float* oh_o   = prob_o + (size_t)B_ * TDEC * V_;

  // ---- weight preparation ----
  k_cast<<<cdiv((long)V_ * E_, 256), 256, 0, stream>>>(emb16, emb, V_ * E_);
  for (int l = 0; l < 2; ++l)
    for (int d = 0; d < 2; ++d) {
      int din = (l == 0) ? E_ : DH_;
      k_cast<<<cdiv((long)G_ * din, 256), 256, 0, stream>>>(ewih16[l][d], enc_wih[l][d], G_ * din);
      k_cast<<<cdiv((long)G_ * H_, 256), 256, 0, stream>>>(ewhh16[l][d], enc_whh[l][d], G_ * H_);
      k_bias_sum<<<cdiv(G_, 256), 256, 0, stream>>>(ebsum[l][d], enc_bih[l][d], enc_bhh[l][d], G_);
    }
  for (int l = 0; l < 2; ++l) {
    int din = (l == 0) ? E_ : DH_;
    k_cast<<<cdiv((long)GD_ * din, 256), 256, 0, stream>>>(dwih16[l], dec_wih[l], GD_ * din);
    k_cast<<<cdiv((long)GD_ * DH_, 256), 256, 0, stream>>>(dwhh16[l], dec_whh[l], GD_ * DH_);
    k_bias_sum<<<cdiv(GD_, 256), 256, 0, stream>>>(dbsum[l], dec_bih[l], dec_bhh[l], GD_);
  }
  k_zero_f16<<<cdiv((long)HEADP * 1024, 256), 256, 0, stream>>>(headw16, HEADP * 1024);
  k_cast<<<cdiv((long)103 * 1024, 256), 256, 0, stream>>>(headw16, head_w, 103 * 1024);
  for (int i = 0; i < 3; ++i) {
    k_zero_f16<<<cdiv((long)hp[i] * 1024, 256), 256, 0, stream>>>(projw16[i], hp[i] * 1024);
    k_cast<<<cdiv((long)hsz[i] * 1024, 256), 256, 0, stream>>>(projw16[i], proj_w[i], hsz[i] * 1024);
    k_zero_f16<<<cdiv((long)oszp[i] * Kp[i], 256), 256, 0, stream>>>(outw16[i], oszp[i] * Kp[i]);
    k_copy2d_cast<<<cdiv((long)osz[i] * hsz[i], 256), 256, 0, stream>>>(
        outw16[i], out_w[i], osz[i], hsz[i], Kp[i], hsz[i]);
    k_zero_f16<<<cdiv((long)B_ * Kp[i], 256), 256, 0, stream>>>(projf16[i], B_ * Kp[i]);
  }

  // ---- encoder ----
  k_embed_seq<<<cdiv((long)L_ * B_ * E_, 256), 256, 0, stream>>>(x, emb16, xs0);
  for (int d = 0; d < 2; ++d) {  // layer 0
    k_wmma_gemm_nt<<<dim3(G_ / 16, (L_ * B_) / 16), 32, 0, stream>>>(
        xs0, E_, ewih16[0][d], ebsum[0][d], Gbuf, G_, E_);
    k_enc_recur<<<1, 512, 0, stream>>>(Gbuf, ewhh16[0][d], xs1, h0init, c0st, d, d * H_);
  }
  for (int d = 0; d < 2; ++d) {  // layer 1
    k_wmma_gemm_nt<<<dim3(G_ / 16, (L_ * B_) / 16), 32, 0, stream>>>(
        xs1, DH_, ewih16[1][d], ebsum[1][d], Gbuf, G_, DH_);
    k_enc_recur<<<1, 512, 0, stream>>>(Gbuf, ewhh16[1][d], xs2, h1init, c1st, d, d * H_);
  }

  // ---- decoder init ----
  k_dec_init<<<1, 32, 0, stream>>>(start_sym, ys_o, tok);
  k_cast<<<cdiv((long)B_ * DH_, 256), 256, 0, stream>>>(h0f16[0], h0init, B_ * DH_);
  k_cast<<<cdiv((long)B_ * DH_, 256), 256, 0, stream>>>(h1f16[0], h1init, B_ * DH_);

  // ---- autoregressive decode ----
  for (int t = 0; t < TDEC; ++t) {
    int pin = t & 1, pout = pin ^ 1;
    k_embed_tok<<<cdiv((long)B_ * E_, 256), 256, 0, stream>>>(tok, emb16, e16);
    k_dec_cell<<<1, 1024, 0, stream>>>(e16, E_, h0f16[pin], dwih16[0], dwhh16[0],
                                       dbsum[0], c0st, hstep0, h0f16[pout]);
    k_dec_cell<<<1, 1024, 0, stream>>>(h0f16[pout], DH_, h1f16[pin], dwih16[1], dwhh16[1],
                                       dbsum[1], c1st, hstep1, h1f16[pout]);
    k_attn<<<B_, 256, 0, stream>>>(xs2, hstep1, attn_b);
    k_ctx<<<B_, 256, 0, stream>>>(xs2, attn_b, hstep1, lg16);
    k_critic<<<B_, 128, 0, stream>>>(hstep1, crit_w, crit_b, crit_o, t);
    k_wmma_gemm_nt<<<dim3(HEADP / 16, 2), 32, 0, stream>>>(
        lg16, 1024, headw16, nullptr, headlog, HEADP, 1024);
    k_head_lse<<<B_, 128, 0, stream>>>(headlog, headls);
    k_head_probs<<<B_, 128, 0, stream>>>(headls, prob_o, cand_v, cand_i, t);
    for (int i = 0; i < 3; ++i) {
      k_wmma_gemm_nt<<<dim3(hp[i] / 16, 2), 32, 0, stream>>>(
          lg16, 1024, projw16[i], nullptr, projf32, Kp[i], 1024);
      k_copy2d_cast<<<cdiv((long)B_ * hp[i], 256), 256, 0, stream>>>(
          projf16[i], projf32, B_, hp[i], Kp[i], Kp[i]);
      k_wmma_gemm_nt<<<dim3(oszp[i] / 16, 2), 32, 0, stream>>>(
          projf16[i], Kp[i], outw16[i], nullptr, taillog, oszp[i], Kp[i]);
      k_tail_probs<<<B_, 256, 0, stream>>>(taillog, oszp[i], osz[i], headls, 100 + i,
                                           prob_o, vbase[i], cand_v, cand_i, i + 1, t);
    }
    k_finalize<<<1, 32, 0, stream>>>(cand_v, cand_i, ys_o, val_o, tok, t);
    k_onehot<<<cdiv((long)B_ * V_, 256), 256, 0, stream>>>(tok, oh_o, t);
  }
}
